// GATGAE_28501402977098
// MI455X (gfx1250) — matrix operand order repaired and checked by reading, not compiled, not run on hardware
//
#include <hip/hip_runtime.h>
#include <hip/hip_bf16.h>
#include <math.h>

typedef __attribute__((ext_vector_type(16))) _Float16 v16h;
typedef __attribute__((ext_vector_type(8)))  float    v8f;

#define TPB 256

// ---------- order-preserving float <-> uint map for atomic max ----------
__device__ __forceinline__ unsigned fmax_map(float f) {
  unsigned u = __float_as_uint(f);
  return (u & 0x80000000u) ? ~u : (u | 0x80000000u);
}
__device__ __forceinline__ float fmax_unmap(unsigned u) {
  unsigned b = (u & 0x80000000u) ? (u & 0x7FFFFFFFu) : ~u;
  return __uint_as_float(b);
}

// ---------- utility kernels ----------
__global__ void k_fill_zero(float* p, long long n) {
  long long t = (long long)blockIdx.x * blockDim.x + threadIdx.x;
  if (t < n) p[t] = 0.0f;
}

__global__ void k_edge_norm(const float* __restrict__ ea, const float* g, const float* b,
                            const float* mu, const float* var, float* __restrict__ e, int E) {
  int t = blockIdx.x * blockDim.x + threadIdx.x;
  if (t >= E) return;
  float v = log1pf(ea[t]);
  e[t] = (v - mu[0]) * rsqrtf(var[0] + 1e-5f) * g[0] + b[0];
}

// Kc[0..3] = sum_c We1[h*32+c]*ae1[h*32+c]   (EDGE_DIM==1 collapse)
// Kc[4]    = sum_c We2[c]*ae2[c]
__global__ void k_precompute_k(const float* We1, const float* ae1,
                               const float* We2, const float* ae2, float* Kc) {
  int t = threadIdx.x;
  if (t < 4) {
    float s = 0.f;
    for (int c = 0; c < 32; ++c) s += We1[t * 32 + c] * ae1[t * 32 + c];
    Kc[t] = s;
  } else if (t == 4) {
    float s = 0.f;
    for (int c = 0; c < 32; ++c) s += We2[c] * ae2[c];
    Kc[4] = s;
  }
}

// ---------- pre-pack W[K x NC] (f32 row-major) into WMMA B lane layout (f16) ----------
// Wp[((ct*(K/32) + kt)*32 + lane)*16 + i] = W[(kt*32 + (lane>>4)*16 + i)*NC + ct*16 + (lane&15)]
template <int K, int NC>
__global__ void k_pack_w(const float* __restrict__ W, _Float16* __restrict__ Wp) {
  const int total = (NC / 16) * (K / 32) * 32 * 16;
  int t = blockIdx.x * blockDim.x + threadIdx.x;
  if (t >= total) return;
  int i  = t & 15;
  int l  = (t >> 4) & 31;
  int kt = (t >> 9) % (K / 32);
  int ct = t / (512 * (K / 32));
  int k = kt * 32 + (l >> 4) * 16 + i;
  int n = ct * 16 + (l & 15);
  Wp[t] = (_Float16)W[(long long)k * NC + n];
}

// ---------- WMMA GEMM: out[M x NC] = act(A[M x K] @ W[K x NC] + bias) ----------
// One wave per 16x16 tile; A loaded f32 via float4 (2 aligned 32B runs/lane),
// B loaded as one contiguous 32B f16 vector per lane from the pre-packed buffer.
// Epilogue: uniform full-tile fast path (no per-element EXEC churn).
template <int K, int NC, int ACT>
__global__ void k_wmma_gemm(const float* __restrict__ A, const _Float16* __restrict__ Wp,
                            const float* __restrict__ bias, float* __restrict__ out, int M) {
  const int ntiles = NC / 16;
  int wave = blockIdx.x * (blockDim.x >> 5) + (threadIdx.x >> 5);
  int lane = threadIdx.x & 31;
  int rt = wave / ntiles;
  int ct = wave % ntiles;
  int row0 = rt * 16;
  if (row0 >= M) return;                 // uniform per-wave exit: EXEC stays all-ones
  int col0 = ct * 16;
  int lm = lane & 15;
  int lg = lane >> 4;                    // lane group 0/1
  int row = row0 + lm;
  int rclamp = (row < M) ? row : (M - 1);
  float rmask = (row < M) ? 1.0f : 0.0f;

  v8f acc = {};
#pragma unroll
  for (int kt = 0; kt < K / 32; ++kt) {
    const int kk = kt * 32;
    // A (contiguous k order, same as packed B): lane group 0 -> K 0..15 ; group 1 -> K 16..31
    const float* arow = A + (long long)rclamp * K + kk + lg * 16;
    float4 p0 = ((const float4*)arow)[0];
    float4 p1 = ((const float4*)arow)[1];
    float4 q0 = ((const float4*)(arow + 8))[0];
    float4 q1 = ((const float4*)(arow + 8))[1];
    float av[16] = {p0.x, p0.y, p0.z, p0.w, p1.x, p1.y, p1.z, p1.w,
                    q0.x, q0.y, q0.z, q0.w, q1.x, q1.y, q1.z, q1.w};
    v16h a = {};
#pragma unroll
    for (int i = 0; i < 16; ++i) a[i] = (_Float16)(av[i] * rmask);

    v16h b = *(const v16h*)(Wp + ((size_t)(ct * (K / 32) + kt) * 32 + lane) * 16);

    acc = __builtin_amdgcn_wmma_f32_16x16x32_f16(false, a, false, b,
                                                 (short)0, acc, false, false);
  }
  float bb = bias ? bias[col0 + lm] : 0.0f;
  float* orow = out + (long long)(row0 + lg * 8) * NC + col0 + lm;
  if (row0 + 16 <= M) {
    // uniform fast path: whole 16-row tile in range, no per-element guards
#pragma unroll
    for (int v = 0; v < 8; ++v) {
      float val = acc[v] + bb;
      if (ACT == 1) val = (val > 0.f) ? val : (expf(val) - 1.f);  // ELU
      orow[(long long)v * NC] = val;
    }
  } else {
#pragma unroll
    for (int v = 0; v < 8; ++v) {
      int m = row0 + lg * 8 + v;         // D: VGPR v -> row lg*8+v, col = lm
      if (m < M) {
        float val = acc[v] + bb;
        if (ACT == 1) val = (val > 0.f) ? val : (expf(val) - 1.f);
        orow[(long long)v * NC] = val;
      }
    }
  }
}

// ---------- per-node attention logits ----------
template <int H, int C>
__global__ void k_node_alpha(const float* __restrict__ h, const float* __restrict__ a_src,
                             const float* __restrict__ a_dst, float* __restrict__ asrc,
                             float* __restrict__ adst, int n_nodes) {
  int t = blockIdx.x * blockDim.x + threadIdx.x;
  if (t >= n_nodes * H) return;
  int n = t / H, hh = t % H;
  const float* row = h + (long long)n * (H * C) + hh * C;
  float s1 = 0.f, s2 = 0.f;
#pragma unroll
  for (int c = 0; c < C; ++c) {
    s1 += row[c] * a_src[hh * C + c];
    s2 += row[c] * a_dst[hh * C + c];
  }
  asrc[t] = s1;
  adst[t] = s2;
}

// ---------- per-edge alpha + segment max ----------
template <int H>
__global__ void k_edge_alpha_max(const int* __restrict__ src, const int* __restrict__ dst,
                                 const float* __restrict__ e, const float* __restrict__ Kc,
                                 const float* __restrict__ asrc, const float* __restrict__ adst,
                                 float* __restrict__ alpha, unsigned* __restrict__ amax_u, int E) {
  long long t = (long long)blockIdx.x * blockDim.x + threadIdx.x;
  if (t >= (long long)E * H) return;
  long long i = t / H;
  int hh = (int)(t % H);
  int s = src[i], d = dst[i];
  float a = asrc[(long long)s * H + hh] + adst[(long long)d * H + hh] + e[i] * Kc[hh];
  a = (a > 0.f) ? a : 0.2f * a;          // leaky_relu(0.2)
  alpha[t] = a;
  atomicMax(&amax_u[(long long)d * H + hh], fmax_map(a));
}

// ---------- exp(alpha - amax) + segment sum ----------
template <int H>
__global__ void k_edge_exp_sum(const int* __restrict__ dst, float* __restrict__ alpha,
                               const unsigned* __restrict__ amax_u, float* __restrict__ denom,
                               int E) {
  long long t = (long long)blockIdx.x * blockDim.x + threadIdx.x;
  if (t >= (long long)E * H) return;
  long long i = t / H;
  int hh = (int)(t % H);
  int d = dst[i];
  unsigned u = amax_u[(long long)d * H + hh];
  float am = (u == 0u) ? 0.f : fmax_unmap(u);   // empty segment -> 0 (matches ref)
  float ex = expf(alpha[t] - am);
  alpha[t] = ex;                                 // overwrite with exp
  atomicAdd(&denom[(long long)d * H + hh], ex);
}

// ---------- message scatter-add: agg[dst] += h[src] * att (float4 gathers) ----------
template <int H, int C>
__global__ void k_edge_aggregate(const int* __restrict__ src, const int* __restrict__ dst,
                                 const float* __restrict__ h, const float* __restrict__ ex,
                                 const float* __restrict__ denom, float* __restrict__ agg,
                                 int E) {
  const int CG = C / 4;
  long long t = (long long)blockIdx.x * blockDim.x + threadIdx.x;
  if (t >= (long long)E * H * CG) return;
  int cg = (int)(t % CG);
  long long eh = t / CG;
  int hh = (int)(eh % H);
  long long i = eh / H;
  int s = src[i], d = dst[i];
  float att = ex[(long long)i * H + hh] / (denom[(long long)d * H + hh] + 1e-16f);
  float4 hv = *(const float4*)(h + (long long)s * (H * C) + hh * C + cg * 4);
  float* ap = agg + (long long)d * (H * C) + hh * C + cg * 4;
  atomicAdd(ap + 0, hv.x * att);
  atomicAdd(ap + 1, hv.y * att);
  atomicAdd(ap + 2, hv.z * att);
  atomicAdd(ap + 3, hv.w * att);
}

// ---------- elementwise bias (+ optional ELU) ----------
template <int ACT>
__global__ void k_bias_act(const float* __restrict__ in, const float* __restrict__ bias,
                           float* __restrict__ out, int cols, long long n) {
  long long t = (long long)blockIdx.x * blockDim.x + threadIdx.x;
  if (t >= n) return;
  float v = in[t] + bias[(int)(t % cols)];
  if (ACT == 1) v = (v > 0.f) ? v : (expf(v) - 1.f);
  out[t] = v;
}

static inline unsigned gridFor(long long n) { return (unsigned)((n + TPB - 1) / TPB); }

extern "C" void kernel_launch(void* const* d_in, const int* in_sizes, int n_in,
                              void* d_out, int out_size, void* d_ws, size_t ws_size,
                              hipStream_t stream) {
  const float* x      = (const float*)d_in[0];
  const int*   eidx   = (const int*)d_in[1];
  const float* eattr  = (const float*)d_in[2];
  const float* bng    = (const float*)d_in[3];
  const float* bnb    = (const float*)d_in[4];
  const float* bnm    = (const float*)d_in[5];
  const float* bnv    = (const float*)d_in[6];
  const float* W1     = (const float*)d_in[7];
  const float* as1    = (const float*)d_in[8];
  const float* ad1    = (const float*)d_in[9];
  const float* We1    = (const float*)d_in[10];
  const float* ae1    = (const float*)d_in[11];
  const float* b1     = (const float*)d_in[12];
  const float* W2     = (const float*)d_in[13];
  const float* as2    = (const float*)d_in[14];
  const float* ad2    = (const float*)d_in[15];
  const float* We2    = (const float*)d_in[16];
  const float* ae2    = (const float*)d_in[17];
  const float* b2     = (const float*)d_in[18];
  const float* Wd1    = (const float*)d_in[19];
  const float* bd1    = (const float*)d_in[20];
  const float* Wd2    = (const float*)d_in[21];
  const float* bd2    = (const float*)d_in[22];

  const int N = in_sizes[0] / 64;
  const int E = in_sizes[1] / 2;
  const int* src = eidx;
  const int* dst = eidx + E;

  // ---------- workspace carve-up (float units, 32B-aligned blocks) ----------
  float* ws = (float*)d_ws;
  size_t off = 0;
  auto align8 = [&]() { off = (off + 7) & ~(size_t)7; };
  float* e_norm = ws + off; off += (size_t)E;            align8();
  float* Kc     = ws + off; off += 8;                    align8();
  _Float16* wp1  = (_Float16*)(ws + off); off += 4096;   align8();  // 8*2*512 halves
  _Float16* wp2  = (_Float16*)(ws + off); off += 2048;   align8();  // 2*4*512 halves
  _Float16* wpd1 = (_Float16*)(ws + off); off += 512;    align8();  // 2*1*512 halves
  _Float16* wpd2 = (_Float16*)(ws + off); off += 1024;   align8();  // 4*1*512 halves
  float* h1     = ws + off; off += (size_t)N * 128;      align8();
  float* asrc1  = ws + off; off += (size_t)N * 4;        align8();
  float* adst1  = ws + off; off += (size_t)N * 4;        align8();
  float* alpha1 = ws + off; off += (size_t)E * 4;        align8();
  float* h2     = ws + off; off += (size_t)N * 32;       align8();
  float* asrc2  = ws + off; off += (size_t)N;            align8();
  float* adst2  = ws + off; off += (size_t)N;            align8();
  float* alpha2 = ws + off; off += (size_t)E;            align8();
  float* dec1   = ws + off; off += (size_t)N * 32;       align8();
  // zero-init region (contiguous): amax1, denom1, agg1, amax2, denom2, agg2
  float*    zstart = ws + off;
  unsigned* amax1  = (unsigned*)(ws + off); off += (size_t)N * 4;   align8();
  float*    denom1 = ws + off;              off += (size_t)N * 4;   align8();
  float*    agg1   = ws + off;              off += (size_t)N * 128; align8();
  unsigned* amax2  = (unsigned*)(ws + off); off += (size_t)N;       align8();
  float*    denom2 = ws + off;              off += (size_t)N;       align8();
  float*    agg2   = ws + off;              off += (size_t)N * 32;  align8();
  const long long zcount = (long long)((ws + off) - zstart);

  float* z_out  = (float*)d_out;                  // [N,32]
  float* xh_out = (float*)d_out + (size_t)N * 32; // [N,64]

  // 1) zero accumulators (every call; harness doesn't re-poison between replays)
  k_fill_zero<<<gridFor(zcount), TPB, 0, stream>>>(zstart, zcount);

  // 2) edge normalization, collapsed edge constants, weight pre-packing
  k_edge_norm<<<gridFor(E), TPB, 0, stream>>>(eattr, bng, bnb, bnm, bnv, e_norm, E);
  k_precompute_k<<<1, 64, 0, stream>>>(We1, ae1, We2, ae2, Kc);
  k_pack_w<64, 128><<<gridFor(8 * 2 * 512), TPB, 0, stream>>>(W1, wp1);
  k_pack_w<128, 32><<<gridFor(2 * 4 * 512), TPB, 0, stream>>>(W2, wp2);
  k_pack_w<32, 32><<<gridFor(2 * 1 * 512), TPB, 0, stream>>>(Wd1, wpd1);
  k_pack_w<32, 64><<<gridFor(4 * 1 * 512), TPB, 0, stream>>>(Wd2, wpd2);

  const int mt = (N + 15) / 16;

  // 3) layer-1 GEMM: h1 = x @ W1   (K=64, NC=128)
  k_wmma_gemm<64, 128, 0><<<(mt * 8 + 7) / 8, TPB, 0, stream>>>(x, wp1, nullptr, h1, N);

  // 4) layer-1 attention (H=4, C=32)
  k_node_alpha<4, 32><<<gridFor((long long)N * 4), TPB, 0, stream>>>(h1, as1, ad1, asrc1, adst1, N);
  k_edge_alpha_max<4><<<gridFor((long long)E * 4), TPB, 0, stream>>>(src, dst, e_norm, Kc, asrc1, adst1, alpha1, amax1, E);
  k_edge_exp_sum<4><<<gridFor((long long)E * 4), TPB, 0, stream>>>(dst, alpha1, amax1, denom1, E);
  k_edge_aggregate<4, 32><<<gridFor((long long)E * 32), TPB, 0, stream>>>(src, dst, h1, alpha1, denom1, agg1, E);
  // 5) h1e = elu(agg1 + b1) (in place)
  k_bias_act<1><<<gridFor((long long)N * 128), TPB, 0, stream>>>(agg1, b1, agg1, 128, (long long)N * 128);

  // 6) layer-2 GEMM: h2 = h1e @ W2  (K=128, NC=32)
  k_wmma_gemm<128, 32, 0><<<(mt * 2 + 7) / 8, TPB, 0, stream>>>(agg1, wp2, nullptr, h2, N);

  // 7) layer-2 attention (H=1, C=32)
  k_node_alpha<1, 32><<<gridFor((long long)N), TPB, 0, stream>>>(h2, as2, ad2, asrc2, adst2, N);
  k_edge_alpha_max<1><<<gridFor((long long)E), TPB, 0, stream>>>(src, dst, e_norm, Kc + 4, asrc2, adst2, alpha2, amax2, E);
  k_edge_exp_sum<1><<<gridFor((long long)E), TPB, 0, stream>>>(dst, alpha2, amax2, denom2, E);
  k_edge_aggregate<1, 32><<<gridFor((long long)E * 8), TPB, 0, stream>>>(src, dst, h2, alpha2, denom2, agg2, E);
  // 8) z = agg2 + b2  -> first output
  k_bias_act<0><<<gridFor((long long)N * 32), TPB, 0, stream>>>(agg2, b2, z_out, 32, (long long)N * 32);

  // 9) decoder: x_hat = elu(z @ Wd1 + bd1) @ Wd2 + bd2 -> second output
  k_wmma_gemm<32, 32, 1><<<(mt * 2 + 7) / 8, TPB, 0, stream>>>(z_out, wpd1, bd1, dec1, N);
  k_wmma_gemm<32, 64, 0><<<(mt * 4 + 7) / 8, TPB, 0, stream>>>(dec1, wpd2, bd2, xh_out, N);
}
